// PointNet2Classification_78254304133664
// MI455X (gfx1250) — matrix operand order repaired and checked by reading, not compiled, not run on hardware
//
#include <hip/hip_runtime.h>

typedef __attribute__((ext_vector_type(16))) _Float16 v16h;
typedef __attribute__((ext_vector_type(8)))  _Float16 v8h;
typedef __attribute__((ext_vector_type(8)))  float    v8f;

#define BN_EPS 1e-5f

// ============================================================================
// Farthest point sampling: one block per batch, dist[] lives in LDS.
// ============================================================================
__global__ void fps_kernel(const float* __restrict__ xyz, int N, int npoint,
                           int* __restrict__ fidx, float* __restrict__ newxyz) {
  int b = blockIdx.x;
  int tid = threadIdx.x;
  __shared__ float dist[4096];
  __shared__ float rmax[256];
  __shared__ int   rarg[256];
  __shared__ int   sfar;

  for (int n = tid; n < N; n += 256) dist[n] = 1e10f;
  if (tid == 0) sfar = 0;
  __syncthreads();

  for (int it = 0; it < npoint; ++it) {
    int far = sfar;
    if (tid == 0) {
      fidx[b * npoint + it] = far;
      newxyz[(size_t)(b * npoint + it) * 3 + 0] = xyz[((size_t)b * N + far) * 3 + 0];
      newxyz[(size_t)(b * npoint + it) * 3 + 1] = xyz[((size_t)b * N + far) * 3 + 1];
      newxyz[(size_t)(b * npoint + it) * 3 + 2] = xyz[((size_t)b * N + far) * 3 + 2];
    }
    float cx = xyz[((size_t)b * N + far) * 3 + 0];
    float cy = xyz[((size_t)b * N + far) * 3 + 1];
    float cz = xyz[((size_t)b * N + far) * 3 + 2];

    float best = -1.0f; int barg = 0;
    for (int n = tid; n < N; n += 256) {
      float dx = xyz[((size_t)b * N + n) * 3 + 0] - cx;
      float dy = xyz[((size_t)b * N + n) * 3 + 1] - cy;
      float dz = xyz[((size_t)b * N + n) * 3 + 2] - cz;
      float d = dx * dx + dy * dy + dz * dz;
      float dn = fminf(dist[n], d);
      dist[n] = dn;
      if (dn > best) { best = dn; barg = n; }
    }
    rmax[tid] = best; rarg[tid] = barg;
    __syncthreads();
    for (int st = 128; st > 0; st >>= 1) {
      if (tid < st) {
        if (rmax[tid + st] > rmax[tid] ||
            (rmax[tid + st] == rmax[tid] && rarg[tid + st] < rarg[tid])) {
          rmax[tid] = rmax[tid + st];
          rarg[tid] = rarg[tid + st];
        }
      }
      __syncthreads();
    }
    if (tid == 0) sfar = rarg[0];
    __syncthreads();
  }
}

// ============================================================================
// Ball query: one thread per (b, m); first-`ns` indices in ascending order
// (matches jnp.sort of masked indices). Centroid is always within radius.
// ============================================================================
__global__ void ball_query_kernel(const float* __restrict__ xyz,
                                  const float* __restrict__ new_xyz,
                                  int B, int N, int M, float r2, int ns,
                                  int* __restrict__ idx) {
  int q = blockIdx.x * blockDim.x + threadIdx.x;
  if (q >= B * M) return;
  int b = q / M, m = q % M;
  float cx = new_xyz[(size_t)(b * M + m) * 3 + 0];
  float cy = new_xyz[(size_t)(b * M + m) * 3 + 1];
  float cz = new_xyz[(size_t)(b * M + m) * 3 + 2];
  int cnt = 0, first = 0;
  for (int n = 0; n < N; ++n) {
    float dx = xyz[((size_t)b * N + n) * 3 + 0] - cx;
    float dy = xyz[((size_t)b * N + n) * 3 + 1] - cy;
    float dz = xyz[((size_t)b * N + n) * 3 + 2] - cz;
    float d2 = dx * dx + dy * dy + dz * dz;
    if (d2 < r2) {
      if (cnt == 0) first = n;
      idx[(size_t)q * ns + cnt] = n;
      cnt++;
      if (cnt >= ns) break;
    }
  }
  for (int j = cnt; j < ns; ++j) idx[(size_t)q * ns + j] = first;
}

// ============================================================================
// Gather kernels: build fp16 GEMM operand X[c][s], channel-padded with zeros.
// ============================================================================
__global__ void gather_sa1_kernel(const float* __restrict__ xyz,
                                  const int* __restrict__ idx,
                                  int B, int N, int M, int ns, int Cpad,
                                  _Float16* __restrict__ X) {
  int s = blockIdx.x * blockDim.x + threadIdx.x;
  int S = B * M * ns;
  if (s >= S) return;
  int b = s / (M * ns);
  int id = idx[s];
  for (int c = 0; c < 3; ++c)
    X[(size_t)c * S + s] = (_Float16)xyz[((size_t)b * N + id) * 3 + c];
  for (int c = 3; c < Cpad; ++c)
    X[(size_t)c * S + s] = (_Float16)0.0f;
}

__global__ void gather_sa2_kernel(const float* __restrict__ xyz_src,
                                  const float* __restrict__ new_xyz,
                                  const float* __restrict__ feat,  // [B][Cf][Np]
                                  const int* __restrict__ idx,
                                  int B, int Np, int M, int ns, int Cf, int Cpad,
                                  _Float16* __restrict__ X) {
  int s = blockIdx.x * blockDim.x + threadIdx.x;
  int S = B * M * ns;
  if (s >= S) return;
  int b = s / (M * ns);
  int m = (s / ns) % M;
  int id = idx[s];
  for (int c = 0; c < 3; ++c) {
    float v = xyz_src[((size_t)b * Np + id) * 3 + c] -
              new_xyz[((size_t)b * M + m) * 3 + c];
    X[(size_t)c * S + s] = (_Float16)v;
  }
  for (int c = 0; c < Cf; ++c)
    X[(size_t)(3 + c) * S + s] =
        (_Float16)feat[((size_t)b * Cf + c) * Np + id];
  for (int c = 3 + Cf; c < Cpad; ++c)
    X[(size_t)c * S + s] = (_Float16)0.0f;
}

__global__ void gather_sa3_kernel(const float* __restrict__ xyz2,   // [B][K][3]
                                  const float* __restrict__ f2,     // [B][Cf][K]
                                  int B, int K, int Cf, int Cpad,
                                  _Float16* __restrict__ X) {
  int t = blockIdx.x * blockDim.x + threadIdx.x;
  if (t >= B * K) return;
  int b = t / K, k = t % K;
  int S = B * K;
  int s = t;
  for (int c = 0; c < 3; ++c)
    X[(size_t)c * S + s] = (_Float16)xyz2[((size_t)b * K + k) * 3 + c];
  for (int c = 0; c < Cf; ++c)
    X[(size_t)(3 + c) * S + s] = (_Float16)f2[((size_t)b * Cf + c) * K + k];
  for (int c = 3 + Cf; c < Cpad; ++c)
    X[(size_t)c * S + s] = (_Float16)0.0f;
}

// ============================================================================
// Weight prep: W (Cin,Cout) f32 row-major  ->  Wh[o][c] fp16, K-padded.
// ============================================================================
__global__ void prep_w_kernel(const float* __restrict__ W, int Cin, int Cout,
                              int Cinp, _Float16* __restrict__ Wh) {
  int t = blockIdx.x * blockDim.x + threadIdx.x;
  if (t >= Cout * Cinp) return;
  int o = t / Cinp, c = t % Cinp;
  Wh[t] = (c < Cin) ? (_Float16)W[(size_t)c * Cout + o] : (_Float16)0.0f;
}

// ============================================================================
// WMMA GEMM: Y[o][s] = sum_c Wh[o][c] * X[c][s].
// 128 threads = 4 wave32s; block tile 32(o) x 128(s), K-step 32.
// Wave w owns o-subtile (w&1) and s-subtiles (w>>1)*4 .. +3:
//   1 A-fragment load, 4 B-fragment loads, 4 WMMAs per K-step.
// LDS tiles are stored pre-swizzled in WMMA fragment order so each lane's
// v16h fragment is a contiguous 32B LDS read (2 x ds_load_b128).
// Global staging is vectorized: b128 loads (8 halfs) + ds_store_b16 scatter.
// Fragment maps follow CDNA5 ISA 7.12.2 (16-bit A 16x32, B 32x16, f32 C/D).
// ============================================================================
__global__ __launch_bounds__(128)
void gemm_wmma_kernel(const _Float16* __restrict__ Wh,
                      const _Float16* __restrict__ X,
                      float* __restrict__ Y,
                      int Cinp, int S) {
  int o0 = blockIdx.x * 32;
  int s0 = blockIdx.y * 128;
  int tid  = threadIdx.x;
  int lane = tid & 31;
  int wave = tid >> 5;
  int asub   = wave & 1;          // o subtile (0..1)
  int sgroup = (wave >> 1) * 4;   // first of 4 s subtiles (0..7)

  __shared__ __align__(32) _Float16 Alds[2][512];  // [o-subtile][lane*16+elem]
  __shared__ __align__(32) _Float16 Blds[8][512];  // [s-subtile][lane*16+elem]

  v8f acc[4] = {};

  for (int k0 = 0; k0 < Cinp; k0 += 32) {
    if (k0 + 32 < Cinp) {
      __builtin_prefetch(&Wh[(size_t)o0 * Cinp + (k0 + 32)], 0, 1);
      __builtin_prefetch(&X[(size_t)(k0 + 32) * S + s0], 0, 1);
    }
    // ---- A tile: 32 rows x 32 k = 1024 halfs = 128 x v8h, one per thread.
    {
      int r  = tid >> 2;          // 0..31 (o row)
      int kv = (tid & 3) * 8;     // k group of 8
      v8h av = *(const v8h*)&Wh[(size_t)(o0 + r) * Cinp + (k0 + kv)];
      int sub = r >> 4;
      int rr  = r & 15;
#pragma unroll
      for (int j = 0; j < 8; ++j) {
        int k = kv + j;
        int h = (k >> 3) & 1;
        int e = (k & 7) + 8 * ((k >> 4) & 1);
        Alds[sub][(rr + 16 * h) * 16 + e] = av[j];
      }
    }
    // ---- B tile: 32 k x 128 s = 4096 halfs = 512 x v8h, 4 per thread.
#pragma unroll
    for (int i = 0; i < 4; ++i) {
      int vt = tid + 128 * i;
      int kk = vt >> 4;           // 0..31 (k row)
      int nv = (vt & 15) * 8;     // s group of 8
      v8h bv = *(const v8h*)&X[(size_t)(k0 + kk) * S + (s0 + nv)];
      int h = (kk >> 3) & 1;
      int e = (kk & 7) + 8 * ((kk >> 4) & 1);
#pragma unroll
      for (int j = 0; j < 8; ++j) {
        int n = nv + j;
        Blds[n >> 4][((n & 15) + 16 * h) * 16 + e] = bv[j];
      }
    }
    __syncthreads();

    v16h a = *(const v16h*)&Alds[asub][lane * 16];
#pragma unroll
    for (int j = 0; j < 4; ++j) {
      v16h b = *(const v16h*)&Blds[sgroup + j][lane * 16];
      acc[j] = __builtin_amdgcn_wmma_f32_16x16x32_f16(
          /*neg_a=*/false, a, /*neg_b=*/false, b,
          /*c_mod=*/(short)0, acc[j], /*reuse_a=*/false, /*reuse_b=*/false);
    }
    __syncthreads();
  }

  // D layout (f32 16x16): lane 0-15 -> N=lane, M=v; lane 16-31 -> M=v+8.
  int h = lane >> 4;
  int n = lane & 15;
  int obase = o0 + asub * 16;
#pragma unroll
  for (int j = 0; j < 4; ++j) {
    int sbase = s0 + (sgroup + j) * 16;
#pragma unroll
    for (int v = 0; v < 8; ++v) {
      int m = v + 8 * h;
      Y[(size_t)(obase + m) * S + (sbase + n)] = acc[j][v];
    }
  }
}

// ============================================================================
// BatchNorm statistics per output channel over S (biased var, eps=1e-5).
// ============================================================================
__global__ void bn_stats_kernel(const float* __restrict__ Y, int S,
                                float* __restrict__ mean,
                                float* __restrict__ rstd) {
  int o = blockIdx.x;
  int tid = threadIdx.x;
  __shared__ float s1[256], s2[256];
  float a = 0.0f, b = 0.0f;
  for (int s = tid; s < S; s += 256) {
    float y = Y[(size_t)o * S + s];
    a += y; b += y * y;
  }
  s1[tid] = a; s2[tid] = b;
  __syncthreads();
  for (int st = 128; st > 0; st >>= 1) {
    if (tid < st) { s1[tid] += s1[tid + st]; s2[tid] += s2[tid + st]; }
    __syncthreads();
  }
  if (tid == 0) {
    float m = s1[0] / (float)S;
    float v = s2[0] / (float)S - m * m;
    mean[o] = m;
    rstd[o] = rsqrtf(v + BN_EPS);
  }
}

// ============================================================================
// Normalize + ReLU: writes f32 back into Y and (optionally) fp16 next input.
// ============================================================================
__global__ void bn_relu_kernel(float* __restrict__ Y,
                               const float* __restrict__ mean,
                               const float* __restrict__ rstd,
                               int S, int Cout, _Float16* __restrict__ Xout) {
  size_t i = (size_t)blockIdx.x * 256 + threadIdx.x;
  if (i >= (size_t)Cout * S) return;
  int o = (int)(i / (size_t)S);
  float y = (Y[i] - mean[o]) * rstd[o];
  y = fmaxf(y, 0.0f);
  Y[i] = y;
  if (Xout) Xout[i] = (_Float16)y;
}

// ============================================================================
// Max-pool over neighborhood k: out[b][coff+o][m] = max_j Y[o][(b*M+m)*ns+j]
// ============================================================================
__global__ void maxpool_kernel(const float* __restrict__ Y,
                               int B, int M, int ns, int Cout,
                               int Ctot, int coff, float* __restrict__ out) {
  int i = blockIdx.x * blockDim.x + threadIdx.x;
  if (i >= B * Cout * M) return;
  int m = i % M;
  int o = (i / M) % Cout;
  int b = i / (M * Cout);
  size_t base = (size_t)o * ((size_t)B * M * ns) + (size_t)(b * M + m) * ns;
  float mx = -1e30f;
  for (int j = 0; j < ns; ++j) mx = fmaxf(mx, Y[base + j]);
  out[((size_t)b * Ctot + coff + o) * M + m] = mx;
}

// ============================================================================
// Classifier head: (8,1024) -> 512 -> 256 -> 1, BN over batch axis, sigmoid.
// One block; tiny problem.
// ============================================================================
__global__ void classifier_kernel(const float* __restrict__ x,     // 8x1024
                                  const float* __restrict__ w0, const float* __restrict__ b0,
                                  const float* __restrict__ w1, const float* __restrict__ b1,
                                  const float* __restrict__ w2, const float* __restrict__ b2,
                                  float* __restrict__ out) {
  __shared__ float h1[8 * 512];
  __shared__ float h2[8 * 256];
  int tid = threadIdx.x;

  for (int t = tid; t < 8 * 512; t += 256) {
    int b = t >> 9, f = t & 511;
    float s = b0[f];
    for (int c = 0; c < 1024; ++c) s += x[b * 1024 + c] * w0[c * 512 + f];
    h1[t] = s;
  }
  __syncthreads();
  for (int f = tid; f < 512; f += 256) {
    float m = 0.0f;
    for (int b = 0; b < 8; ++b) m += h1[b * 512 + f];
    m *= 0.125f;
    float v = 0.0f;
    for (int b = 0; b < 8; ++b) { float d = h1[b * 512 + f] - m; v += d * d; }
    v *= 0.125f;
    float rs = rsqrtf(v + BN_EPS);
    for (int b = 0; b < 8; ++b)
      h1[b * 512 + f] = fmaxf((h1[b * 512 + f] - m) * rs, 0.0f);
  }
  __syncthreads();
  for (int t = tid; t < 8 * 256; t += 256) {
    int b = t >> 8, f = t & 255;
    float s = b1[f];
    for (int c = 0; c < 512; ++c) s += h1[b * 512 + c] * w1[c * 256 + f];
    h2[t] = s;
  }
  __syncthreads();
  for (int f = tid; f < 256; f += 256) {
    float m = 0.0f;
    for (int b = 0; b < 8; ++b) m += h2[b * 256 + f];
    m *= 0.125f;
    float v = 0.0f;
    for (int b = 0; b < 8; ++b) { float d = h2[b * 256 + f] - m; v += d * d; }
    v *= 0.125f;
    float rs = rsqrtf(v + BN_EPS);
    for (int b = 0; b < 8; ++b)
      h2[b * 256 + f] = fmaxf((h2[b * 256 + f] - m) * rs, 0.0f);
  }
  __syncthreads();
  if (tid < 8) {
    float s = b2[0];
    for (int c = 0; c < 256; ++c) s += h2[tid * 256 + c] * w2[c];
    out[tid] = 1.0f / (1.0f + expf(-s));
  }
}

// ============================================================================
// Host-side helper: one shared-MLP layer = prep_w + WMMA GEMM + BN + ReLU.
// ============================================================================
static void run_layer(hipStream_t st, const float* W, int Cin, int Cout,
                      const _Float16* Xin, _Float16* Xout, float* Y,
                      _Float16* Wh, float* mean, float* rstd, int S) {
  int Cinp = (Cin + 31) & ~31;
  int nw = Cout * Cinp;
  prep_w_kernel<<<(nw + 255) / 256, 256, 0, st>>>(W, Cin, Cout, Cinp, Wh);
  dim3 g(Cout / 32, S / 128);
  gemm_wmma_kernel<<<g, 128, 0, st>>>(Wh, Xin, Y, Cinp, S);
  bn_stats_kernel<<<Cout, 256, 0, st>>>(Y, S, mean, rstd);
  size_t tot = (size_t)Cout * S;
  bn_relu_kernel<<<(unsigned)((tot + 255) / 256), 256, 0, st>>>(Y, mean, rstd,
                                                               S, Cout, Xout);
}

extern "C" void kernel_launch(void* const* d_in, const int* in_sizes, int n_in,
                              void* d_out, int out_size, void* d_ws, size_t ws_size,
                              hipStream_t stream) {
  (void)in_sizes; (void)n_in; (void)out_size; (void)ws_size;

  const float* pc = (const float*)d_in[0];                 // (8,4096,3)
  const float* sa1w[3][3];
  const float* sa2w[3][3];
  for (int i = 0; i < 3; ++i)
    for (int j = 0; j < 3; ++j) {
      sa1w[i][j] = (const float*)d_in[1 + i * 3 + j];
      sa2w[i][j] = (const float*)d_in[10 + i * 3 + j];
    }
  const float* sa3w[3];
  const float* clsw[3];
  const float* clsb[3];
  for (int l = 0; l < 3; ++l) {
    sa3w[l] = (const float*)d_in[19 + l];
    clsw[l] = (const float*)d_in[22 + l];
    clsb[l] = (const float*)d_in[25 + l];
  }

  // ---- workspace carve (deterministic fixed offsets) ----
  char* wsb = (char*)d_ws;
  size_t off = 0;
  auto carve = [&](size_t bytes) -> void* {
    void* r = wsb + off;
    off += (bytes + 255) & ~(size_t)255;
    return r;
  };
  int*      fidx1 = (int*)carve((size_t)8 * 512 * 4);
  float*    xyz1  = (float*)carve((size_t)8 * 512 * 3 * 4);
  int*      fidx2 = (int*)carve((size_t)8 * 128 * 4);
  float*    xyz2  = (float*)carve((size_t)8 * 128 * 3 * 4);
  float*    f1    = (float*)carve((size_t)8 * 320 * 512 * 4);
  float*    f2    = (float*)carve((size_t)8 * 640 * 128 * 4);
  float*    f3    = (float*)carve((size_t)8 * 1024 * 4);
  int*      idxb  = (int*)carve((size_t)8 * 512 * 128 * 4);
  _Float16* Wh    = (_Float16*)carve((size_t)1024 * 1024 * 2);
  float*    mean  = (float*)carve((size_t)1024 * 4);
  float*    rstd  = (float*)carve((size_t)1024 * 4);
  _Float16* Xa    = (_Float16*)carve((size_t)96 * 1024 * 1024);
  _Float16* Xb    = (_Float16*)carve((size_t)96 * 1024 * 1024);
  float*    Y     = (float*)carve((size_t)256 * 1024 * 1024);

  // ---- Stage SA1 ----
  fps_kernel<<<8, 256, 0, stream>>>(pc, 4096, 512, fidx1, xyz1);
  {
    const float r1[3] = {0.1f, 0.2f, 0.4f};
    const int   ns1[3] = {16, 32, 128};
    const int   dims1[3][4] = {{3, 32, 32, 64}, {3, 64, 64, 128}, {3, 64, 64, 128}};
    const int   coff1[3] = {0, 64, 192};
    for (int br = 0; br < 3; ++br) {
      int M = 512, Nn = 4096, ns = ns1[br];
      int Q = 8 * M;
      ball_query_kernel<<<(Q + 127) / 128, 128, 0, stream>>>(
          pc, xyz1, 8, Nn, M, r1[br] * r1[br], ns, idxb);
      int S = 8 * M * ns;
      gather_sa1_kernel<<<(S + 255) / 256, 256, 0, stream>>>(
          pc, idxb, 8, Nn, M, ns, 32, Xa);
      const _Float16* Xin = Xa;
      _Float16* Xout = Xb;
      for (int l = 0; l < 3; ++l) {
        _Float16* xo = (l == 2) ? nullptr : Xout;
        run_layer(stream, sa1w[br][l], dims1[br][l], dims1[br][l + 1],
                  Xin, xo, Y, Wh, mean, rstd, S);
        if (l < 2) { const _Float16* t = Xin; Xin = Xout; Xout = (_Float16*)t; }
      }
      int Cl = dims1[br][3];
      int tot = 8 * Cl * M;
      maxpool_kernel<<<(tot + 255) / 256, 256, 0, stream>>>(
          Y, 8, M, ns, Cl, 320, coff1[br], f1);
    }
  }

  // ---- Stage SA2 ----
  fps_kernel<<<8, 256, 0, stream>>>(xyz1, 512, 128, fidx2, xyz2);
  {
    const float r2a[3] = {0.2f, 0.4f, 0.8f};
    const int   ns2[3] = {32, 64, 128};
    const int   dims2[3][4] = {{323, 64, 64, 128}, {323, 128, 128, 256}, {323, 128, 128, 256}};
    const int   coff2[3] = {0, 128, 384};
    for (int br = 0; br < 3; ++br) {
      int M = 128, Np = 512, ns = ns2[br];
      int Q = 8 * M;
      ball_query_kernel<<<(Q + 127) / 128, 128, 0, stream>>>(
          xyz1, xyz2, 8, Np, M, r2a[br] * r2a[br], ns, idxb);
      int S = 8 * M * ns;
      gather_sa2_kernel<<<(S + 255) / 256, 256, 0, stream>>>(
          xyz1, xyz2, f1, idxb, 8, Np, M, ns, 320, 352, Xa);
      const _Float16* Xin = Xa;
      _Float16* Xout = Xb;
      for (int l = 0; l < 3; ++l) {
        _Float16* xo = (l == 2) ? nullptr : Xout;
        run_layer(stream, sa2w[br][l], dims2[br][l], dims2[br][l + 1],
                  Xin, xo, Y, Wh, mean, rstd, S);
        if (l < 2) { const _Float16* t = Xin; Xin = Xout; Xout = (_Float16*)t; }
      }
      int Cl = dims2[br][3];
      int tot = 8 * Cl * M;
      maxpool_kernel<<<(tot + 255) / 256, 256, 0, stream>>>(
          Y, 8, M, ns, Cl, 640, coff2[br], f2);
    }
  }

  // ---- Stage SA3 (global aggregation over 128 points, M=1) ----
  {
    int S = 8 * 128;
    gather_sa3_kernel<<<(8 * 128 + 255) / 256, 256, 0, stream>>>(
        xyz2, f2, 8, 128, 640, 672, Xa);
    const int dims3[4] = {643, 256, 512, 1024};
    const _Float16* Xin = Xa;
    _Float16* Xout = Xb;
    for (int l = 0; l < 3; ++l) {
      _Float16* xo = (l == 2) ? nullptr : Xout;
      run_layer(stream, sa3w[l], dims3[l], dims3[l + 1],
                Xin, xo, Y, Wh, mean, rstd, S);
      if (l < 2) { const _Float16* t = Xin; Xin = Xout; Xout = (_Float16*)t; }
    }
    int tot = 8 * 1024;
    maxpool_kernel<<<(tot + 255) / 256, 256, 0, stream>>>(
        Y, 8, 1, 128, 1024, 1024, 0, f3);
  }

  // ---- Classifier head ----
  classifier_kernel<<<1, 256, 0, stream>>>(f3, clsw[0], clsb[0], clsw[1],
                                           clsb[1], clsw[2], clsb[2],
                                           (float*)d_out);
}
